// Attn_88364657148092
// MI455X (gfx1250) — compile-verified, matrix operand order checked by reading, not run on hardware
//
#include <hip/hip_runtime.h>
#include <math.h>

#define B_   2
#define T_   2048
#define D_   1024
#define NH_  16
#define NKV_ 4
#define HD_  64
#define BT_  (B_*T_)

typedef unsigned short u16;
typedef __attribute__((ext_vector_type(16))) __bf16        v16bf;
typedef __attribute__((ext_vector_type(8)))  float          v8f;
typedef __attribute__((ext_vector_type(8)))  unsigned short u16x8;

// log2(10000 * 2^(32/31)) for NTK-scaled RoPE base (T=2048 > TSL=1024)
#define LOG2NB 14.3199704f

__device__ __forceinline__ u16 f2bf(float f) {
  unsigned u = __float_as_uint(f);
  return (u16)((u + 0x7FFFu + ((u >> 16) & 1u)) >> 16);
}

// Load one 16-bit WMMA fragment (A or B) for lane-half h.
// ISA layout: half h holds K = {8h..8h+7} and {16+8h..23+8h} -> two 16B loads.
__device__ __forceinline__ v16bf ldfrag(const u16* p, int h) {
  union { u16x8 u[2]; v16bf v; } cv;
  cv.u[0] = *(const u16x8*)(p + 8*h);
  cv.u[1] = *(const u16x8*)(p + 16 + 8*h);
  return cv.v;
}

__device__ __forceinline__ v8f wmma_bf16(v16bf a, v16bf b, v8f c) {
  return __builtin_amdgcn_wmma_f32_16x16x32_bf16(false, a, false, b,
                                                 (short)0, c, false, false);
}

__device__ __forceinline__ v8f v8zero() {
  v8f z;
#pragma unroll
  for (int i = 0; i < 8; ++i) z[i] = 0.0f;
  return z;
}

// ---------------------------------------------------------------- convert
__global__ __launch_bounds__(256) void k_convert(
    const float* __restrict__ x,  const float* __restrict__ qw,
    const float* __restrict__ kw, const float* __restrict__ vw,
    const float* __restrict__ ow,
    u16* __restrict__ xb, u16* __restrict__ wqkvb, u16* __restrict__ owb) {
  const int NX = BT_ * D_;            // 4096*1024
  const int NW = 1536 * D_;           // qkv weights concat
  const int NO = D_ * D_;
  int stride = gridDim.x * blockDim.x;
  for (int i = blockIdx.x * blockDim.x + threadIdx.x; i < NX + NW + NO; i += stride) {
    if (i < NX) {
      xb[i] = f2bf(x[i]);
    } else if (i < NX + NW) {
      int j = i - NX;
      float v;
      if (j < 1024 * D_)        v = qw[j];
      else if (j < 1280 * D_)   v = kw[j - 1024 * D_];
      else                      v = vw[j - 1280 * D_];
      wqkvb[j] = f2bf(v);
    } else {
      int j = i - NX - NW;
      owb[j] = f2bf(ow[j]);
    }
  }
}

// --------------------------------------------------------------- WMMA GEMM
// C[M][N] = A[M][K] * W[N][K]^T.
// Register-blocked: each wave owns a 16x64 output strip (4 accumulators),
// so per K-step: 1 A-frag + 4 B-frags feed 4 WMMAs (2.5 b128 loads / wmma).
__global__ __launch_bounds__(256) void k_gemm(
    const u16* __restrict__ A, const u16* __restrict__ W,
    float* __restrict__ C, int N, int K) {
  int lane = threadIdx.x & 31;
  int warp = threadIdx.x >> 5;
  int h    = lane >> 4;
  int l16  = lane & 15;
  int m0 = blockIdx.x * 16;
  int n0 = blockIdx.y * 512 + warp * 64;
  const u16* arow = A + (size_t)(m0 + l16) * K;
  const u16* w0 = W + (size_t)(n0      + l16) * K;
  const u16* w1 = W + (size_t)(n0 + 16 + l16) * K;
  const u16* w2 = W + (size_t)(n0 + 32 + l16) * K;
  const u16* w3 = W + (size_t)(n0 + 48 + l16) * K;
  v8f c0 = v8zero(), c1 = v8zero(), c2 = v8zero(), c3 = v8zero();
#pragma unroll 2
  for (int k0 = 0; k0 < K; k0 += 32) {
    v16bf a = ldfrag(arow + k0, h);
    c0 = wmma_bf16(a, ldfrag(w0 + k0, h), c0);
    c1 = wmma_bf16(a, ldfrag(w1 + k0, h), c1);
    c2 = wmma_bf16(a, ldfrag(w2 + k0, h), c2);
    c3 = wmma_bf16(a, ldfrag(w3 + k0, h), c3);
  }
#pragma unroll
  for (int r = 0; r < 8; ++r) {
    int row = m0 + r + 8 * h;            // C layout: VGPR r -> M=r (+8 for hi half)
    float* crow = C + (size_t)row * N + n0 + l16;
    crow[0]  = c0[r];
    crow[16] = c1[r];
    crow[32] = c2[r];
    crow[48] = c3[r];
  }
}

// ---------------------------------------------------------------- epilogue
// RMS-norm + RoPE (+gain) for Q/K, v-residual + raw_v + transposed V store.
__global__ __launch_bounds__(256) void k_epilogue(
    const float* __restrict__ qkvf, const float* __restrict__ ve,
    const float* __restrict__ v0,   const float* __restrict__ qgain,
    const float* __restrict__ vlam,
    u16* __restrict__ qb, u16* __restrict__ kb, u16* __restrict__ vt,
    float* __restrict__ rawv_out) {
  int bt   = blockIdx.x;
  int b    = bt >> 11;
  int t    = bt & 2047;
  int warp = threadIdx.x >> 5;
  int lane = threadIdx.x & 31;
  const float* row = qkvf + (size_t)bt * 1536;

  float inv = exp2f(-((float)(2 * lane) * (1.0f / 64.0f)) * LOG2NB);
  float ang = (float)t * inv;
  float cs = cosf(ang), sn = sinf(ang);

  // ---- Q heads: two passes of 8 warps, each wave owns one head
#pragma unroll
  for (int pass = 0; pass < 2; ++pass) {
    int qh = pass * 8 + warp;
    const float* hp = row + qh * 64;
    float v1 = hp[lane], v2 = hp[lane + 32];
    float ss = v1 * v1 + v2 * v2;
    ss += __shfl_xor(ss, 1, 32);  ss += __shfl_xor(ss, 2, 32);
    ss += __shfl_xor(ss, 4, 32);  ss += __shfl_xor(ss, 8, 32);
    ss += __shfl_xor(ss, 16, 32);
    float rms = rsqrtf(ss * (1.0f / 64.0f) + 1.1920929e-07f);
    float n1 = v1 * rms, n2 = v2 * rms;
    float g  = qgain[qh];
    float o1 = ( n1 * cs + n2 * sn) * g;
    float o2 = (-n1 * sn + n2 * cs) * g;
    u16* dst = qb + (((size_t)(b * NH_ + qh)) * T_ + t) * 64;
    dst[lane]      = f2bf(o1);
    dst[lane + 32] = f2bf(o2);
  }

  if (warp < 4) {
    // ---- K heads
    int kh = warp;
    const float* hp = row + 1024 + kh * 64;
    float v1 = hp[lane], v2 = hp[lane + 32];
    float ss = v1 * v1 + v2 * v2;
    ss += __shfl_xor(ss, 1, 32);  ss += __shfl_xor(ss, 2, 32);
    ss += __shfl_xor(ss, 4, 32);  ss += __shfl_xor(ss, 8, 32);
    ss += __shfl_xor(ss, 16, 32);
    float rms = rsqrtf(ss * (1.0f / 64.0f) + 1.1920929e-07f);
    float n1 = v1 * rms, n2 = v2 * rms;
    float o1 =  n1 * cs + n2 * sn;
    float o2 = -n1 * sn + n2 * cs;
    u16* dst = kb + (((size_t)(b * NKV_ + kh)) * T_ + t) * 64;
    dst[lane]      = f2bf(o1);
    dst[lane + 32] = f2bf(o2);
  } else {
    // ---- V: raw_v out + lambda-mix + transposed bf16 store
    float l0 = vlam[0], l1 = vlam[1];
    int base = (warp - 4) * 64 + lane;
#pragma unroll
    for (int off = 0; off < 64; off += 32) {
      int c = base + off;
      float raw = row[1280 + c] + ve[(size_t)bt * 256 + c];
      rawv_out[(size_t)bt * 256 + c] = raw;
      float vm = l0 * v0[(size_t)bt * 256 + c] + l1 * raw;
      int hh = c >> 6, d = c & 63;
      vt[(((size_t)(b * NKV_ + hh)) * 64 + d) * T_ + t] = f2bf(vm);
    }
  }
}

// -------------------------------------------------------------------- gate
__global__ __launch_bounds__(256) void k_gate(
    const float* __restrict__ x, const float* __restrict__ gw,
    const float* __restrict__ gb, float* __restrict__ gate) {
  int bt = blockIdx.x;
  int warp = threadIdx.x >> 5, lane = threadIdx.x & 31;
  const float* xr = x + (size_t)bt * D_;
#pragma unroll
  for (int j = 0; j < 2; ++j) {
    int hh = warp * 2 + j;
    const float* w = gw + (size_t)hh * D_;
    float s = 0.0f;
    for (int k = lane; k < D_; k += 32) s += xr[k] * w[k];
    s += __shfl_xor(s, 1, 32);  s += __shfl_xor(s, 2, 32);
    s += __shfl_xor(s, 4, 32);  s += __shfl_xor(s, 8, 32);
    s += __shfl_xor(s, 16, 32);
    if (lane == 0)
      gate[(size_t)bt * NH_ + hh] = 1.0f / (1.0f + expf(-(s + gb[hh])));
  }
}

// --------------------------------------------------------- flash attention
// One wave per (b, q-head, 16-query tile); 32 keys per iteration.
__global__ __launch_bounds__(256) void k_attn(
    const u16* __restrict__ qb, const u16* __restrict__ kb,
    const u16* __restrict__ vt, const float* __restrict__ gate,
    u16* __restrict__ yb) {
  __shared__ u16 plds[8 * 512];            // per-wave 16x32 bf16 P tile
  int warp = threadIdx.x >> 5, lane = threadIdx.x & 31;
  int h = lane >> 4, l16 = lane & 15;
  int w   = blockIdx.x * 8 + warp;
  int qt  = w & 127;
  int qh  = (w >> 7) & 15;
  int b   = w >> 11;
  int hkv = qh >> 2;                        // GQA: 4 q-heads per kv head
  int q0  = qt * 16;

  const u16* qrow = qb + (((size_t)(b * NH_ + qh)) * T_ + q0 + l16) * 64;
  v16bf qa0 = ldfrag(qrow, h);
  v16bf qa1 = ldfrag(qrow + 32, h);
  const u16* kbase = kb + ((size_t)(b * NKV_ + hkv)) * T_ * 64;
  const u16* vbase = vt + ((size_t)(b * NKV_ + hkv)) * 64 * T_;
  u16* pw = plds + warp * 512;

  v8f o0 = v8zero(), o1 = v8zero(), o2 = v8zero(), o3 = v8zero();
  float mr[8], lr[8];
#pragma unroll
  for (int r = 0; r < 8; ++r) { mr[r] = -INFINITY; lr[r] = 0.0f; }

  for (int j0 = 0; j0 <= q0 + 15; j0 += 32) {
    // ---- S = Q * K^T for 32 keys (two 16x16 tiles, K=64 -> 2 steps each)
    const u16* kcol0 = kbase + (size_t)(j0 + l16) * 64;
    const u16* kcol1 = kbase + (size_t)(j0 + 16 + l16) * 64;
    v8f s0 = v8zero(), s1 = v8zero();
    s0 = wmma_bf16(qa0, ldfrag(kcol0, h), s0);
    s0 = wmma_bf16(qa1, ldfrag(kcol0 + 32, h), s0);
    s1 = wmma_bf16(qa0, ldfrag(kcol1, h), s1);
    s1 = wmma_bf16(qa1, ldfrag(kcol1 + 32, h), s1);

    // ---- online softmax (rows live in 16-lane halves of the C layout)
    float al[8];
#pragma unroll
    for (int r = 0; r < 8; ++r) {
      int rowm = q0 + r + 8 * h;
      float a0 = s0[r] * 0.125f;
      float a1 = s1[r] * 0.125f;
      if (j0 + l16 > rowm)      a0 = -INFINITY;
      if (j0 + 16 + l16 > rowm) a1 = -INFINITY;
      float mx = fmaxf(a0, a1);
      mx = fmaxf(mx, __shfl_xor(mx, 1, 32));
      mx = fmaxf(mx, __shfl_xor(mx, 2, 32));
      mx = fmaxf(mx, __shfl_xor(mx, 4, 32));
      mx = fmaxf(mx, __shfl_xor(mx, 8, 32));
      float mnew  = fmaxf(mr[r], mx);
      float alpha = expf(mr[r] - mnew);
      float p0 = expf(a0 - mnew), p1 = expf(a1 - mnew);
      float ps = p0 + p1;
      ps += __shfl_xor(ps, 1, 32);  ps += __shfl_xor(ps, 2, 32);
      ps += __shfl_xor(ps, 4, 32);  ps += __shfl_xor(ps, 8, 32);
      lr[r] = lr[r] * alpha + ps;
      mr[r] = mnew;
      al[r] = alpha;
      int rr = r + 8 * h;
      pw[rr * 32 + l16]      = f2bf(p0);
      pw[rr * 32 + 16 + l16] = f2bf(p1);
    }
#pragma unroll
    for (int r = 0; r < 8; ++r) {
      o0[r] *= al[r]; o1[r] *= al[r]; o2[r] *= al[r]; o3[r] *= al[r];
    }

    // ---- O += P * V (P re-laid-out through LDS into A-fragment form)
    v16bf pa = ldfrag(pw + l16 * 32, h);
    o0 = wmma_bf16(pa, ldfrag(vbase + (size_t)(l16)      * T_ + j0, h), o0);
    o1 = wmma_bf16(pa, ldfrag(vbase + (size_t)(16 + l16) * T_ + j0, h), o1);
    o2 = wmma_bf16(pa, ldfrag(vbase + (size_t)(32 + l16) * T_ + j0, h), o2);
    o3 = wmma_bf16(pa, ldfrag(vbase + (size_t)(48 + l16) * T_ + j0, h), o3);
  }

  // ---- normalize, gate, store bf16 y
#pragma unroll
  for (int r = 0; r < 8; ++r) {
    int rowm = q0 + r + 8 * h;
    float g = gate[((size_t)(b * T_ + rowm)) * NH_ + qh];
    float inv = g / lr[r];
    u16* dst = yb + ((size_t)(b * T_ + rowm)) * D_ + qh * 64;
    dst[l16]      = f2bf(o0[r] * inv);
    dst[16 + l16] = f2bf(o1[r] * inv);
    dst[32 + l16] = f2bf(o2[r] * inv);
    dst[48 + l16] = f2bf(o3[r] * inv);
  }
}

// ------------------------------------------------------------------ launch
extern "C" void kernel_launch(void* const* d_in, const int* in_sizes, int n_in,
                              void* d_out, int out_size, void* d_ws, size_t ws_size,
                              hipStream_t stream) {
  const float* x  = (const float*)d_in[0];
  const float* qw = (const float*)d_in[1];
  const float* kw = (const float*)d_in[2];
  const float* vw = (const float*)d_in[3];
  const float* ow = (const float*)d_in[4];
  const float* ve = (const float*)d_in[5];
  const float* v0 = (const float*)d_in[6];
  const float* qg = (const float*)d_in[7];
  const float* vl = (const float*)d_in[8];
  const float* gw = (const float*)d_in[9];
  const float* gb = (const float*)d_in[10];
  float* out = (float*)d_out;

  char* ws = (char*)d_ws;
  size_t off = 0;
  auto carve = [&](size_t bytes) {
    void* p = ws + off;
    off += (bytes + 255) & ~(size_t)255;
    return p;
  };
  u16*   xb    = (u16*)  carve((size_t)BT_ * D_ * 2);
  u16*   wqkvb = (u16*)  carve((size_t)1536 * D_ * 2);
  u16*   owb   = (u16*)  carve((size_t)D_ * D_ * 2);
  u16*   qb    = (u16*)  carve((size_t)BT_ * D_ * 2);
  u16*   kbuf  = (u16*)  carve((size_t)BT_ * 256 * 2);
  u16*   vt    = (u16*)  carve((size_t)BT_ * 256 * 2);
  u16*   yb    = (u16*)  carve((size_t)BT_ * D_ * 2);
  float* qkvf  = (float*)carve((size_t)BT_ * 1536 * 4);
  float* gate  = (float*)carve((size_t)BT_ * NH_ * 4);
  (void)ws_size; (void)in_sizes; (void)n_in; (void)out_size;

  float* rawv_out = out + (size_t)BT_ * D_;   // second tuple element

  hipLaunchKernelGGL(k_convert, dim3(2048), dim3(256), 0, stream,
                     x, qw, kw, vw, ow, xb, wqkvb, owb);
  hipLaunchKernelGGL(k_gemm, dim3(BT_ / 16, 3), dim3(256), 0, stream,
                     xb, wqkvb, qkvf, 1536, D_);
  hipLaunchKernelGGL(k_epilogue, dim3(BT_), dim3(256), 0, stream,
                     qkvf, ve, v0, qg, vl, qb, kbuf, vt, rawv_out);
  hipLaunchKernelGGL(k_gate, dim3(BT_), dim3(256), 0, stream,
                     x, gw, gb, gate);
  hipLaunchKernelGGL(k_attn, dim3((B_ * NH_ * (T_ / 16)) / 8), dim3(256), 0, stream,
                     qb, kbuf, vt, gate, yb);
  hipLaunchKernelGGL(k_gemm, dim3(BT_ / 16, 2), dim3(256), 0, stream,
                     yb, owb, out, D_, D_);
}
